// BinaryTreeLSTMCell_88519275971251
// MI455X (gfx1250) — matrix operand order repaired and checked by reading, not compiled, not run on hardware
//
#include <hip/hip_runtime.h>

// ---------------------------------------------------------------------------
// BinaryTreeLSTM cell, fused, bf16 WMMA (f32 accum) for gfx1250 / MI455X.
//
// GEMM formulation (K = 256 via concatenation):
//   iou  = [x | (l_h+r_h)] @ [[W_ioux],[W_iouh]] + (b_ioux+b_iouh)   [B,384]
//   f_l  = sigmoid([x | l_h] @ [[W_fx],[W_fh]] + (b_fx+b_fh))        [B,128]
//   f_r  = sigmoid([x | r_h] @ [[W_fx],[W_fh]] + (b_fx+b_fh))        [B,128]
//   c = sig(i)*tanh(u) + f_l*l_c + f_r*r_c ;  h = sig(o)*tanh(c)
//
// Weights live bf16-transposed in LDS ([N][K=256], row stride 512B) so a
// B-fragment (lane = column, K = 8*hi+0..7 and 16+8*hi+0..7) is exactly two
// ds_load_b128. A-fragments are built straight from global f32 loads using
// the documented 16x32 bf16 A swizzle, then cvt'd to bf16.
// ---------------------------------------------------------------------------

typedef __bf16 bf16_t;
typedef __attribute__((ext_vector_type(16))) __bf16 v16bf;
typedef __attribute__((ext_vector_type(8)))  __bf16 v8bf;
typedef __attribute__((ext_vector_type(8)))  float  v8f;

#define B_TOTAL   262144
#define ROW_TILES (B_TOTAL / 16)   // 16384
#define TPB       256              // 8 wave32
#define GRID      512              // 4096 waves -> 4 tiles/wave

__device__ __forceinline__ float fast_sigmoid(float x) {
  // 1/(1+e^-x); exp2 underflow/overflow saturate correctly via rcp.
  return __builtin_amdgcn_rcpf(1.0f + __builtin_amdgcn_exp2f(-1.4426950408889634f * x));
}
__device__ __forceinline__ float fast_tanh(float x) {
  // 1 - 2/(e^{2x}+1); safe at +-inf of exp2.
  return 1.0f - 2.0f * __builtin_amdgcn_rcpf(1.0f + __builtin_amdgcn_exp2f(2.8853900817779268f * x));
}

__device__ __forceinline__ v16bf ld_bfrag(const bf16_t* p) {
  union { v16bf v; v8bf h[2]; } u;
  u.h[0] = *(const v8bf*)p;        // K = 8*hi + 0..7
  u.h[1] = *(const v8bf*)(p + 16); // K = 16 + 8*hi + 0..7
  return u.v;
}

__device__ __forceinline__ v16bf pack_a(float4 a0, float4 a1, float4 b0, float4 b1) {
  v16bf v;
  v[0]=(bf16_t)a0.x;  v[1]=(bf16_t)a0.y;  v[2]=(bf16_t)a0.z;  v[3]=(bf16_t)a0.w;
  v[4]=(bf16_t)a1.x;  v[5]=(bf16_t)a1.y;  v[6]=(bf16_t)a1.z;  v[7]=(bf16_t)a1.w;
  v[8]=(bf16_t)b0.x;  v[9]=(bf16_t)b0.y;  v[10]=(bf16_t)b0.z; v[11]=(bf16_t)b0.w;
  v[12]=(bf16_t)b1.x; v[13]=(bf16_t)b1.y; v[14]=(bf16_t)b1.z; v[15]=(bf16_t)b1.w;
  return v;
}

#define WMMA_BF16(A, Bm, C) \
  __builtin_amdgcn_wmma_f32_16x16x32_bf16(false, (A), false, (Bm), (short)0, (C), false, false)

__global__ __launch_bounds__(TPB) void treelstm_kernel(
    const float* __restrict__ l_h, const float* __restrict__ l_c,
    const float* __restrict__ r_h, const float* __restrict__ r_c,
    const float* __restrict__ x,
    const float* __restrict__ W_ioux, const float* __restrict__ b_ioux,
    const float* __restrict__ W_iouh, const float* __restrict__ b_iouh,
    const float* __restrict__ W_fx,   const float* __restrict__ b_fx,
    const float* __restrict__ W_fh,   const float* __restrict__ b_fh,
    float* __restrict__ out)
{
  // 264,192 B of LDS (ISA: up to 320 KB / workgroup on CDNA5)
  __shared__ __align__(64) bf16_t sW1[384 * 256];  // [n][k] : k<128 ioux, k>=128 iouh
  __shared__ __align__(64) bf16_t sW2[128 * 256];  // [n][k] : k<128 fx,   k>=128 fh
  __shared__ float sBiou[384];
  __shared__ float sBf[128];

  const int tid = threadIdx.x;

  // -------- one-time per-block weight staging (f32 -> bf16, transpose) -----
  for (int idx = tid; idx < 128 * 384; idx += TPB) {
    int k = idx / 384, n = idx - k * 384;          // source row-major [k][n]
    sW1[n * 256 + k]       = (bf16_t)W_ioux[idx];
    sW1[n * 256 + 128 + k] = (bf16_t)W_iouh[idx];
  }
  for (int idx = tid; idx < 128 * 128; idx += TPB) {
    int k = idx >> 7, n = idx & 127;
    sW2[n * 256 + k]       = (bf16_t)W_fx[idx];
    sW2[n * 256 + 128 + k] = (bf16_t)W_fh[idx];
  }
  for (int n = tid; n < 384; n += TPB) sBiou[n] = b_ioux[n] + b_iouh[n];
  for (int n = tid; n < 128; n += TPB) sBf[n]   = b_fx[n]   + b_fh[n];
  __syncthreads();

  const int lane = tid & 31;
  const int hi   = lane >> 4;       // half-wave select
  const int col  = lane & 15;       // B/C/D column, A row
  const int gw   = blockIdx.x * (TPB / 32) + (tid >> 5);
  const int nw   = gridDim.x * (TPB / 32);

  float* __restrict__ out_h = out;
  float* __restrict__ out_c = out + (size_t)B_TOTAL * 128;

  for (int tile = gw; tile < ROW_TILES; tile += nw) {
    const size_t rbase = (size_t)tile * 16 * 128;
    const float* xp = x   + rbase + (size_t)col * 128;
    const float* lp = l_h + rbase + (size_t)col * 128;
    const float* rp = r_h + rbase + (size_t)col * 128;

    // ---- A fragments: x, l, r, s=l+r (bf16, documented 16x32 swizzle) ----
    v16bf xf[4], lf[4], rf[4], sf[4];
#pragma unroll
    for (int s = 0; s < 4; ++s) {
      const int c0 = 32 * s + 8 * hi;     // K = c0..c0+7 and c0+16..c0+23
      float4 xa0 = *(const float4*)(xp + c0);
      float4 xa1 = *(const float4*)(xp + c0 + 4);
      float4 xb0 = *(const float4*)(xp + c0 + 16);
      float4 xb1 = *(const float4*)(xp + c0 + 20);
      xf[s] = pack_a(xa0, xa1, xb0, xb1);

      float4 la0 = *(const float4*)(lp + c0);
      float4 la1 = *(const float4*)(lp + c0 + 4);
      float4 lb0 = *(const float4*)(lp + c0 + 16);
      float4 lb1 = *(const float4*)(lp + c0 + 20);
      lf[s] = pack_a(la0, la1, lb0, lb1);

      float4 ra0 = *(const float4*)(rp + c0);
      float4 ra1 = *(const float4*)(rp + c0 + 4);
      float4 rb0 = *(const float4*)(rp + c0 + 16);
      float4 rb1 = *(const float4*)(rp + c0 + 20);
      rf[s] = pack_a(ra0, ra1, rb0, rb1);

      float4 sa0, sa1, sb0, sb1;   // f32 sum, then single bf16 rounding
      sa0.x=la0.x+ra0.x; sa0.y=la0.y+ra0.y; sa0.z=la0.z+ra0.z; sa0.w=la0.w+ra0.w;
      sa1.x=la1.x+ra1.x; sa1.y=la1.y+ra1.y; sa1.z=la1.z+ra1.z; sa1.w=la1.w+ra1.w;
      sb0.x=lb0.x+rb0.x; sb0.y=lb0.y+rb0.y; sb0.z=lb0.z+rb0.z; sb0.w=lb0.w+rb0.w;
      sb1.x=lb1.x+rb1.x; sb1.y=lb1.y+rb1.y; sb1.z=lb1.z+rb1.z; sb1.w=lb1.w+rb1.w;
      sf[s] = pack_a(sa0, sa1, sb0, sb1);
    }

    // ---- 8 output column tiles x 8 K-steps x 5 WMMAs ----
    for (int n = 0; n < 8; ++n) {
      const int n0 = 16 * n;
      const float bi = sBiou[n0 + col];
      const float bo = sBiou[128 + n0 + col];
      const float bu = sBiou[256 + n0 + col];
      const float bg = sBf[n0 + col];
      v8f ci  = {bi, bi, bi, bi, bi, bi, bi, bi};
      v8f co  = {bo, bo, bo, bo, bo, bo, bo, bo};
      v8f cu  = {bu, bu, bu, bu, bu, bu, bu, bu};
      v8f cfl = {bg, bg, bg, bg, bg, bg, bg, bg};
      v8f cfr = {bg, bg, bg, bg, bg, bg, bg, bg};

      const bf16_t* w1i = sW1 + (n0 + col) * 256 + 8 * hi;
      const bf16_t* w1o = sW1 + (128 + n0 + col) * 256 + 8 * hi;
      const bf16_t* w1u = sW1 + (256 + n0 + col) * 256 + 8 * hi;
      const bf16_t* w2f = sW2 + (n0 + col) * 256 + 8 * hi;

#pragma unroll
      for (int k = 0; k < 8; ++k) {
        const v16bf a_iou = (k < 4) ? xf[k] : sf[k - 4];
        const v16bf a_l   = (k < 4) ? xf[k] : lf[k - 4];
        const v16bf a_r   = (k < 4) ? xf[k] : rf[k - 4];
        const v16bf Bi = ld_bfrag(w1i + 32 * k);
        const v16bf Bo = ld_bfrag(w1o + 32 * k);
        const v16bf Bu = ld_bfrag(w1u + 32 * k);
        const v16bf Bf = ld_bfrag(w2f + 32 * k);
        ci  = WMMA_BF16(a_iou, Bi, ci);
        co  = WMMA_BF16(a_iou, Bo, co);
        cu  = WMMA_BF16(a_iou, Bu, cu);
        cfl = WMMA_BF16(a_l,   Bf, cfl);
        cfr = WMMA_BF16(a_r,   Bf, cfr);
      }

      // ---- gates + cell update; C/D layout: row = e + 8*hi, col = lane&15
      const size_t ebase = rbase + (size_t)(n0 + col);
      const float* lcp = l_c + ebase;
      const float* rcp = r_c + ebase;
#pragma unroll
      for (int e = 0; e < 8; ++e) {
        const int row = e + 8 * hi;
        const float iv  = fast_sigmoid(ci[e]);
        const float ov  = fast_sigmoid(co[e]);
        const float uv  = fast_tanh(cu[e]);
        const float flv = fast_sigmoid(cfl[e]);
        const float frv = fast_sigmoid(cfr[e]);
        const float lcv = lcp[(size_t)row * 128];
        const float rcv = rcp[(size_t)row * 128];
        const float cv  = iv * uv + flv * lcv + frv * rcv;
        const float hv  = ov * fast_tanh(cv);
        __builtin_nontemporal_store(hv, out_h + ebase + (size_t)row * 128);
        __builtin_nontemporal_store(cv, out_c + ebase + (size_t)row * 128);
      }
    }
  }
}

extern "C" void kernel_launch(void* const* d_in, const int* in_sizes, int n_in,
                              void* d_out, int out_size, void* d_ws, size_t ws_size,
                              hipStream_t stream) {
  (void)in_sizes; (void)n_in; (void)out_size; (void)d_ws; (void)ws_size;
  const float* l_h    = (const float*)d_in[0];
  const float* l_c    = (const float*)d_in[1];
  const float* r_h    = (const float*)d_in[2];
  const float* r_c    = (const float*)d_in[3];
  const float* xin    = (const float*)d_in[4];
  const float* W_ioux = (const float*)d_in[5];
  const float* b_ioux = (const float*)d_in[6];
  const float* W_iouh = (const float*)d_in[7];
  const float* b_iouh = (const float*)d_in[8];
  const float* W_fx   = (const float*)d_in[9];
  const float* b_fx   = (const float*)d_in[10];
  const float* W_fh   = (const float*)d_in[11];
  const float* b_fh   = (const float*)d_in[12];

  treelstm_kernel<<<GRID, TPB, 0, stream>>>(
      l_h, l_c, r_h, r_c, xin,
      W_ioux, b_ioux, W_iouh, b_iouh,
      W_fx, b_fx, W_fh, b_fh,
      (float*)d_out);
}
